// VectorQuantizer_90666759618813
// MI455X (gfx1250) — matrix operand order repaired and checked by reading, not compiled, not run on hardware
//
#include <hip/hip_runtime.h>
#include <hip/hip_bf16.h>
#include <math.h>

typedef __attribute__((ext_vector_type(16))) __bf16 v16bf;
typedef __attribute__((ext_vector_type(8)))  float  v8f;

union FragB { v16bf v; uint4 q[2]; };

#define NUM_CODES 4096
#define DIM       256
#define N_ROWS    32768   // 32 * 32 * 32
#define CHW       262144  // 256*32*32
#define TOTAL_EL  8388608 // 32*256*32*32

// ---------------- codebook: split to bf16 hi/lo + row norms ----------------
__global__ void vq_prep_codebook(const float* __restrict__ cb,
                                 __bf16* __restrict__ cb_hi,
                                 __bf16* __restrict__ cb_lo,
                                 float* __restrict__ ysq) {
  __shared__ float red[256];
  int row = blockIdx.x;
  int t = threadIdx.x;
  float v = cb[(size_t)row * DIM + t];
  __bf16 h = (__bf16)v;
  __bf16 l = (__bf16)(v - (float)h);
  cb_hi[(size_t)row * DIM + t] = h;
  cb_lo[(size_t)row * DIM + t] = l;
  red[t] = v * v;
  __syncthreads();
  for (int s = 128; s > 0; s >>= 1) { if (t < s) red[t] += red[t + s]; __syncthreads(); }
  if (t == 0) ysq[row] = red[0];
}

// ---------------- x: NCHW -> row-major [n][c] bf16 hi/lo (LDS transpose) ----
__global__ void vq_prep_x(const float* __restrict__ in,
                          __bf16* __restrict__ xhi, __bf16* __restrict__ xlo) {
  __shared__ __align__(16) __bf16 th[32][256];
  __shared__ __align__(16) __bf16 tl[32][256];
  int b = blockIdx.x >> 5, h = blockIdx.x & 31;
  int t = threadIdx.x;
  int w = t & 31;
  int cg = t >> 5;                 // 0..7
  const float* src = in + (size_t)b * CHW + h * 32 + w;
  for (int c = cg; c < 256; c += 8) {
    float v = src[(size_t)c * 1024];
    __bf16 hh = (__bf16)v;
    th[w][c] = hh;
    tl[w][c] = (__bf16)(v - (float)hh);
  }
  __syncthreads();
  size_t n0 = (size_t)b * 1024 + h * 32;
  for (int j = 0; j < 4; ++j) {
    int u = t + j * 256;
    int r = u >> 5, q = u & 31;    // 32 rows x 32 uint4 per row
    ((uint4*)(xhi + (n0 + r) * DIM))[q] = ((const uint4*)&th[r][0])[q];
    ((uint4*)(xlo + (n0 + r) * DIM))[q] = ((const uint4*)&tl[r][0])[q];
  }
}

// ------------- main: WMMA bf16x3 distances + running argmin -----------------
// Three independent accumulator chains (hi*hi, hi*lo, lo*hi) so the XDL pipe
// never stalls on a single serial WMMA RAW chain.
__global__ void __launch_bounds__(256) vq_argmin(
    const __bf16* __restrict__ xhi, const __bf16* __restrict__ xlo,
    const __bf16* __restrict__ cbhi, const __bf16* __restrict__ cblo,
    const float* __restrict__ ysq,
    int* __restrict__ idx_out, float* __restrict__ idxf_out) {
  int t = threadIdx.x;
  int wave = t >> 5;
  int lane = t & 31;
  int lr   = lane & 15;
  int half = lane >> 4;
  int rowbase = blockIdx.x * 128 + wave * 16;
  int r = rowbase + lr;

  // A fragments (loop-invariant): 8 k-slices of 32, hi and lo
  FragB ahi[8], alo[8];
  const uint4* xh = (const uint4*)(xhi + (size_t)r * DIM);
  const uint4* xl = (const uint4*)(xlo + (size_t)r * DIM);
#pragma unroll
  for (int ks = 0; ks < 8; ++ks) {
    ahi[ks].q[0] = xh[ks * 4 + half];
    ahi[ks].q[1] = xh[ks * 4 + 2 + half];
    alo[ks].q[0] = xl[ks * 4 + half];
    alo[ks].q[1] = xl[ks * 4 + 2 + half];
  }

  float minv[8]; int mini[8];
#pragma unroll
  for (int i = 0; i < 8; ++i) { minv[i] = 3.4e38f; mini[i] = 0; }

  for (int tile = 0; tile < NUM_CODES / 16; ++tile) {
    int crow = tile * 16 + lr;                     // this lane's code column
    const uint4* bh = (const uint4*)(cbhi + (size_t)crow * DIM);
    const uint4* bl = (const uint4*)(cblo + (size_t)crow * DIM);
    v8f acc_hh = {0.f, 0.f, 0.f, 0.f, 0.f, 0.f, 0.f, 0.f};
    v8f acc_hl = {0.f, 0.f, 0.f, 0.f, 0.f, 0.f, 0.f, 0.f};
    v8f acc_lh = {0.f, 0.f, 0.f, 0.f, 0.f, 0.f, 0.f, 0.f};
#pragma unroll
    for (int ks = 0; ks < 8; ++ks) {
      FragB bhi, blo;
      // B layout: lane holds K = half*8 .. half*8+7 and 16+half*8 .. 16+half*8+7
      bhi.q[0] = bh[ks * 4 + half];
      bhi.q[1] = bh[ks * 4 + 2 + half];
      blo.q[0] = bl[ks * 4 + half];
      blo.q[1] = bl[ks * 4 + 2 + half];
      acc_hh = __builtin_amdgcn_wmma_f32_16x16x32_bf16(false, ahi[ks].v, false, bhi.v,
                                                       (short)0, acc_hh, false, false);
      acc_hl = __builtin_amdgcn_wmma_f32_16x16x32_bf16(false, ahi[ks].v, false, blo.v,
                                                       (short)0, acc_hl, false, false);
      acc_lh = __builtin_amdgcn_wmma_f32_16x16x32_bf16(false, alo[ks].v, false, bhi.v,
                                                       (short)0, acc_lh, false, false);
    }
    float ys = ysq[crow];
#pragma unroll
    for (int i = 0; i < 8; ++i) {
      float dot = acc_hh[i] + acc_hl[i] + acc_lh[i];
      float d = ys - 2.0f * dot;                   // |y|^2 - 2 x.y (x^2 const per row)
      if (d < minv[i]) { minv[i] = d; mini[i] = crow; }
    }
  }

  // reduce across the 16 lanes of each half (the N direction), tie -> lowest idx
#pragma unroll
  for (int i = 0; i < 8; ++i) {
    float mv = minv[i]; int mi = mini[i];
#pragma unroll
    for (int off = 1; off < 16; off <<= 1) {
      float ov = __shfl_xor(mv, off, 32);
      int   oi = __shfl_xor(mi, off, 32);
      if (ov < mv || (ov == mv && oi < mi)) { mv = ov; mi = oi; }
    }
    minv[i] = mv; mini[i] = mi;
  }
  if (lr == 0) {
#pragma unroll
    for (int i = 0; i < 8; ++i) {
      int row = rowbase + i + 8 * half;            // C-fragment: VGPR i -> M = i + 8*half
      idx_out[row] = mini[i];
      idxf_out[row] = (float)mini[i];
    }
  }
}

// ---------------- one-hot encodings (fused zero+scatter) + histogram --------
__global__ void vq_onehot(const int* __restrict__ idx, float* __restrict__ enc,
                          unsigned int* __restrict__ counts) {
  int n = blockIdx.x;
  int k = idx[n];
  int t = threadIdx.x;
  float2* row = (float2*)(enc + (size_t)n * NUM_CODES);   // 8-byte aligned region
#pragma unroll
  for (int q = 0; q < 8; ++q) {
    int c = t * 16 + q * 2;
    float2 z;
    z.x = (c == k) ? 1.0f : 0.0f;
    z.y = (c + 1 == k) ? 1.0f : 0.0f;
    row[t * 8 + q] = z;
  }
  if (t == 0) atomicAdd(&counts[k], 1u);
}

// ---------------- gather quantized (STE) back to NCHW + loss reduction ------
__global__ void vq_quantize_loss(const float* __restrict__ in, const float* __restrict__ cb,
                                 const int* __restrict__ idx,
                                 float* __restrict__ outq, float* __restrict__ lossacc) {
  __shared__ float red[256];
  int b = blockIdx.x >> 5, h = blockIdx.x & 31;
  int t = threadIdx.x; int w = t & 31; int cg = t >> 5;
  int n = b * 1024 + h * 32 + w;
  int k = idx[n];
  const float* crow = cb + (size_t)k * DIM;
  size_t base = (size_t)b * CHW + h * 32 + w;
  float s = 0.f;
  for (int c = cg; c < 256; c += 8) {
    float qv = crow[c];
    size_t a = base + (size_t)c * 1024;
    float x = in[a];
    float d = qv - x;
    outq[a] = x + d;                 // straight-through: x + (q - x), matches ref rounding
    s += d * d;
  }
  red[t] = s; __syncthreads();
  for (int st = 128; st > 0; st >>= 1) { if (t < st) red[t] += red[t + st]; __syncthreads(); }
  if (t == 0) atomicAdd(lossacc, red[0]);
}

// ---------------- scalars: loss + perplexity --------------------------------
__global__ void vq_finalize(const unsigned int* __restrict__ counts,
                            const float* __restrict__ lossacc,
                            float* __restrict__ loss_out, float* __restrict__ perp_out) {
  __shared__ float red[256];
  int t = threadIdx.x;
  float s = 0.f;
  for (int j = t; j < NUM_CODES; j += 256) {
    float p = (float)counts[j] * (1.0f / 32768.0f);
    s += p * logf(p + 1e-10f);
  }
  red[t] = s; __syncthreads();
  for (int st = 128; st > 0; st >>= 1) { if (t < st) red[t] += red[t + st]; __syncthreads(); }
  if (t == 0) {
    *perp_out = expf(-red[0]);
    *loss_out = lossacc[0] * 1.25f / (float)TOTAL_EL;  // (1 + commitment 0.25) * mean
  }
}

__global__ void vq_zero(unsigned int* __restrict__ counts, float* __restrict__ lossacc) {
  int t = threadIdx.x;
  for (int j = t; j < NUM_CODES; j += 256) counts[j] = 0u;
  if (t == 0) *lossacc = 0.f;
}

extern "C" void kernel_launch(void* const* d_in, const int* in_sizes, int n_in,
                              void* d_out, int out_size, void* d_ws, size_t ws_size,
                              hipStream_t stream) {
  (void)in_sizes; (void)n_in; (void)out_size; (void)ws_size;
  const float* inputs   = (const float*)d_in[0];   // [32,256,32,32] fp32
  const float* codebook = (const float*)d_in[1];   // [4096,256] fp32
  float* out = (float*)d_out;

  // d_out layout (floats): quantized | loss | perplexity | encodings | indices
  float* out_quant = out;                                   // 8,388,608
  float* out_loss  = out + 8388608;
  float* out_perp  = out + 8388609;
  float* out_enc   = out + 8388610;                         // 134,217,728
  float* out_idxf  = out + 8388610 + (size_t)N_ROWS * NUM_CODES;

  // workspace carve (bytes)
  char* ws = (char*)d_ws;
  __bf16* cb_hi = (__bf16*)(ws);                            // 2 MB
  __bf16* cb_lo = (__bf16*)(ws + (2u  << 20));              // 2 MB
  __bf16* x_hi  = (__bf16*)(ws + (4u  << 20));              // 16 MB
  __bf16* x_lo  = (__bf16*)(ws + (20u << 20));              // 16 MB
  float*  ysq   = (float*) (ws + (36u << 20));              // 16 KB
  int*    idx   = (int*)   (ws + (36u << 20) + (16u << 10));// 128 KB
  unsigned int* counts = (unsigned int*)(ws + (36u << 20) + (144u << 10)); // 16 KB
  float*  lossacc      = (float*)(ws + (36u << 20) + (160u << 10));

  vq_zero<<<1, 256, 0, stream>>>(counts, lossacc);
  vq_prep_codebook<<<NUM_CODES, 256, 0, stream>>>(codebook, cb_hi, cb_lo, ysq);
  vq_prep_x<<<1024, 256, 0, stream>>>(inputs, x_hi, x_lo);
  vq_argmin<<<N_ROWS / 128, 256, 0, stream>>>(x_hi, x_lo, cb_hi, cb_lo, ysq, idx, out_idxf);
  vq_onehot<<<N_ROWS, 256, 0, stream>>>(idx, out_enc, counts);
  vq_quantize_loss<<<1024, 256, 0, stream>>>(inputs, codebook, idx, out_quant, lossacc);
  vq_finalize<<<1, 256, 0, stream>>>(counts, lossacc, out_loss, out_perp);
}
